// GResBlockMeanConv_35716948034105
// MI455X (gfx1250) — compile-verified, matrix operand order checked by reading, not run on hardware
//
#include <hip/hip_runtime.h>

typedef __attribute__((ext_vector_type(16))) _Float16 v16h;
typedef __attribute__((ext_vector_type(8)))  _Float16 v8h;
typedef __attribute__((ext_vector_type(8)))  float    v8f;

#define NNODES 50000
#define NEDGES 800000
#define DIM 128
#define NEG_SLOPE 0.2f

// ---------------------------------------------------------------------------
// utility
// ---------------------------------------------------------------------------
__global__ void fill_kernel(float* __restrict__ p, float v, int n) {
  int i = blockIdx.x * blockDim.x + threadIdx.x;
  if (i < n) p[i] = v;
}

__global__ void f32_to_f16_kernel(const float* __restrict__ s, _Float16* __restrict__ d, int n) {
  int i = blockIdx.x * blockDim.x + threadIdx.x;
  if (i < n) d[i] = (_Float16)s[i];
}

__device__ inline void edge_sd(const int* __restrict__ ei, int e, int E, int& s, int& d) {
  if (e < E) { s = ei[e]; d = ei[E + e]; }
  else       { s = e - E; d = e - E; }          // self loop
}

__device__ inline float leaky(float v) { return v >= 0.f ? v : NEG_SLOPE * v; }

// float atomic max via sign-split int/uint atomics (init must be -inf)
__device__ inline void atomicMaxF(float* addr, float v) {
  if (v >= 0.f) atomicMax((int*)addr, __float_as_int(v));
  else          atomicMin((unsigned int*)addr, __float_as_uint(v));
}

// ---------------------------------------------------------------------------
// H[nRows x 128] = X[nRows x 128] @ W^T   (X,W pre-converted f16, row-major)
// one 16x16 tile per wave; 8 waves/block -> 128 output cols; K=128 in 4 WMMAs.
// The 4KB A tile (shared by all 8 waves) is staged in LDS with CDNA5
// async global->LDS copies (ASYNCcnt), then read via ds_load_b128.
// ---------------------------------------------------------------------------
__global__ __launch_bounds__(256) void gemm128_wmma(const _Float16* __restrict__ X,
                                                    const _Float16* __restrict__ W,
                                                    float* __restrict__ H) {
  __shared__ _Float16 As[16 * DIM];
  const int tid   = threadIdx.x;
  const int lane  = tid & 31;
  const int wave  = tid >> 5;                // 0..7 -> N tile
  const int mBase = blockIdx.x * 16;

  // cooperative async copy: 256 threads x 16B = 16 rows x 128 f16 = 4 KB
  {
    const int row   = tid >> 4;              // 0..15
    const int chunk = tid & 15;              // 16B chunks within a row
    const _Float16* gp = X + (size_t)(mBase + row) * DIM + chunk * 8;
    unsigned ldsOff = (unsigned)(size_t)(const void*)(As + row * DIM + chunk * 8);
    unsigned long long ga = (unsigned long long)(size_t)gp;
    asm volatile("global_load_async_to_lds_b128 %0, %1, off"
                 :: "v"(ldsOff), "v"(ga) : "memory");
  }
  asm volatile("s_wait_asynccnt 0" ::: "memory");
  __syncthreads();

  const int r  = lane & 15;                  // A row / B col within tile
  const int kh = (lane >> 4) * 8;            // K half select: 0 or 8
  const _Float16* arow = As + r * DIM + kh;
  const _Float16* brow = W + (size_t)(wave * 16 + r) * DIM + kh;

  v8f acc = {};
#pragma unroll
  for (int kb = 0; kb < DIM; kb += 32) {
    v8h alo = *(const v8h*)(arow + kb);
    v8h ahi = *(const v8h*)(arow + kb + 16);
    v8h blo = *(const v8h*)(brow + kb);
    v8h bhi = *(const v8h*)(brow + kb + 16);
    v16h a = __builtin_shufflevector(alo, ahi, 0,1,2,3,4,5,6,7,8,9,10,11,12,13,14,15);
    v16h b = __builtin_shufflevector(blo, bhi, 0,1,2,3,4,5,6,7,8,9,10,11,12,13,14,15);
    acc = __builtin_amdgcn_wmma_f32_16x16x32_f16(false, a, false, b,
                                                 (short)0, acc, false, false);
  }

  // C/D layout: lane<16 -> N=lane, M=v ; lane>=16 -> N=lane-16, M=v+8
  const int mOff = (lane >> 4) * 8;
  float* out = H + (size_t)(mBase + mOff) * DIM + wave * 16 + r;
#pragma unroll
  for (int v = 0; v < 8; ++v) out[(size_t)v * DIM] = acc[v];
}

// ---------------------------------------------------------------------------
// alpha_src[n,h] = <h[n,h,:], a_src[h,:]>, same for dst
// ---------------------------------------------------------------------------
__global__ void alpha_kernel(const float* __restrict__ H,
                             const float* __restrict__ a_src,
                             const float* __restrict__ a_dst,
                             float* __restrict__ asrc, float* __restrict__ adst,
                             int n, int heads, int ch) {
  int i = blockIdx.x * blockDim.x + threadIdx.x;     // node*heads + h
  if (i >= n * heads) return;
  int node = i / heads, h = i - node * heads;
  const float* hp = H + (size_t)node * DIM + h * ch;
  const float* ws = a_src + h * ch;
  const float* wd = a_dst + h * ch;
  float s = 0.f, d = 0.f;
  for (int c = 0; c < ch; ++c) { float v = hp[c]; s += v * ws[c]; d += v * wd[c]; }
  asrc[i] = s; adst[i] = d;
}

// ---------------------------------------------------------------------------
// pass 1: segment max of leaky_relu(asrc[src]+adst[dst]) over dst
// ---------------------------------------------------------------------------
__global__ void edge_max_kernel(const int* __restrict__ ei,
                                const float* __restrict__ asrc,
                                const float* __restrict__ adst,
                                float* __restrict__ emax,
                                int E, int n, int heads) {
  int i = blockIdx.x * blockDim.x + threadIdx.x;
  if (i >= (E + n) * heads) return;
  int e = i / heads, h = i - e * heads;
  int s, d; edge_sd(ei, e, E, s, d);
  float v = leaky(asrc[s * heads + h] + adst[d * heads + h]);
  atomicMaxF(&emax[d * heads + h], v);
}

// ---------------------------------------------------------------------------
// pass 2: ee = exp(e - emax[dst]);  denom[dst] += ee
// ---------------------------------------------------------------------------
__global__ void edge_expsum_kernel(const int* __restrict__ ei,
                                   const float* __restrict__ asrc,
                                   const float* __restrict__ adst,
                                   const float* __restrict__ emax,
                                   float* __restrict__ ee, float* __restrict__ denom,
                                   int E, int n, int heads) {
  int i = blockIdx.x * blockDim.x + threadIdx.x;
  if (i >= (E + n) * heads) return;
  int e = i / heads, h = i - e * heads;
  int s, d; edge_sd(ei, e, E, s, d);
  float v = leaky(asrc[s * heads + h] + adst[d * heads + h]);
  float x = __expf(v - emax[d * heads + h]);
  ee[i] = x;
  atomicAdd(&denom[d * heads + h], x);
}

// ---------------------------------------------------------------------------
// pass 3: out[dst] += h[src] * (ee/denom[dst]);  one wave per edge, 4 ch/lane
// ---------------------------------------------------------------------------
__global__ void edge_aggregate_kernel(const int* __restrict__ ei,
                                      const float* __restrict__ H,
                                      const float* __restrict__ ee,
                                      const float* __restrict__ denom,
                                      float* __restrict__ out,
                                      int E, int n, int ch, int heads) {
  long long t = (long long)blockIdx.x * blockDim.x + threadIdx.x;
  if (t >= (long long)(E + n) * 32) return;
  int e = (int)(t >> 5);
  int lane = (int)(t & 31);
  int c = lane * 4;                // heads*ch == 128, chunk never crosses a head
  int h = c / ch;
  int s, d; edge_sd(ei, e, E, s, d);
  float a = ee[(size_t)e * heads + h] / denom[d * heads + h];
  const float* hs = H + (size_t)s * DIM + c;
  float* op = out + (size_t)d * DIM + c;
#pragma unroll
  for (int i = 0; i < 4; ++i) atomicAdd(&op[i], hs[i] * a);
}

// layer-1 epilogue: bias + relu, emit f16 (only consumer is the next GEMM)
__global__ void bias_relu_f16_kernel(const float* __restrict__ buf,
                                     const float* __restrict__ bias,
                                     _Float16* __restrict__ dst, int n) {
  int i = blockIdx.x * blockDim.x + threadIdx.x;
  if (i >= n) return;
  float v = buf[i] + bias[i & (DIM - 1)];
  dst[i] = (_Float16)fmaxf(v, 0.f);
}

// layer-2 epilogue: bias only, in-place f32
__global__ void bias_kernel(float* __restrict__ buf, const float* __restrict__ bias, int n) {
  int i = blockIdx.x * blockDim.x + threadIdx.x;
  if (i >= n) return;
  buf[i] = buf[i] + bias[i & (DIM - 1)];
}

// ---------------------------------------------------------------------------
// mean conv (raw edges, no self loops)
// ---------------------------------------------------------------------------
__global__ void edge_cnt_kernel(const int* __restrict__ ei, float* __restrict__ cnt, int E) {
  int e = blockIdx.x * blockDim.x + threadIdx.x;
  if (e >= E) return;
  atomicAdd(&cnt[ei[E + e]], 1.f);
}

__global__ void edge_sum_kernel(const int* __restrict__ ei, const float* __restrict__ H,
                                float* __restrict__ s, int E) {
  long long t = (long long)blockIdx.x * blockDim.x + threadIdx.x;
  if (t >= (long long)E * 32) return;
  int e = (int)(t >> 5);
  int c = (int)(t & 31) * 4;
  int sn = ei[e], d = ei[E + e];
  const float* hp = H + (size_t)sn * DIM + c;
  float* op = s + (size_t)d * DIM + c;
#pragma unroll
  for (int i = 0; i < 4; ++i) atomicAdd(&op[i], hp[i]);
}

__global__ void final_kernel(const float* __restrict__ s, const float* __restrict__ cnt,
                             const float* __restrict__ x, float* __restrict__ out, int n) {
  int i = blockIdx.x * blockDim.x + threadIdx.x;
  if (i >= n * DIM) return;
  float cv = cnt[i / DIM];
  float m = cv > 0.f ? s[i] / cv : 0.f;
  out[i] = fmaxf(m + x[i], 0.f);
}

// ---------------------------------------------------------------------------
static inline int cdiv(long long a, long long b) { return (int)((a + b - 1) / b); }

extern "C" void kernel_launch(void* const* d_in, const int* in_sizes, int n_in,
                              void* d_out, int out_size, void* d_ws, size_t ws_size,
                              hipStream_t stream) {
  (void)in_sizes; (void)n_in; (void)out_size; (void)ws_size;
  const float* x   = (const float*)d_in[0];
  const int*   ei  = (const int*)d_in[1];
  const float* W1  = (const float*)d_in[2];
  const float* as1 = (const float*)d_in[3];
  const float* ad1 = (const float*)d_in[4];
  const float* b1  = (const float*)d_in[5];
  const float* W2  = (const float*)d_in[6];
  const float* as2 = (const float*)d_in[7];
  const float* ad2 = (const float*)d_in[8];
  const float* b2  = (const float*)d_in[9];
  float* out = (float*)d_out;

  float* A     = (float*)d_ws;                       // N*128 (h buffer, f32)
  float* B     = A + (size_t)NNODES * DIM;           // N*128 (agg buffer, f32)
  float* asrc  = B + (size_t)NNODES * DIM;           // N*2
  float* adst  = asrc + (size_t)NNODES * 2;          // N*2
  float* emax  = adst + (size_t)NNODES * 2;          // N*2
  float* denom = emax + (size_t)NNODES * 2;          // N*2
  float* ee    = denom + (size_t)NNODES * 2;         // (E+N)*2
  float* cnt   = ee + (size_t)(NEDGES + NNODES) * 2; // N
  _Float16* Xh  = (_Float16*)(cnt + NNODES);         // N*128 f16 GEMM input
  _Float16* W1h = Xh + (size_t)NNODES * DIM;         // 128*128 f16
  _Float16* W2h = W1h + DIM * DIM;                   // 128*128 f16

  const int EL = NEDGES + NNODES;                    // edges incl. self loops
  const int T  = 256;
  const float NEG_INF = -__builtin_huge_valf();

  // one-time f16 conversions (removes all in-GEMM cvt redundancy)
  f32_to_f16_kernel<<<cdiv((long long)NNODES * DIM, T), T, 0, stream>>>(x, Xh, NNODES * DIM);
  f32_to_f16_kernel<<<cdiv(DIM * DIM, T), T, 0, stream>>>(W1, W1h, DIM * DIM);
  f32_to_f16_kernel<<<cdiv(DIM * DIM, T), T, 0, stream>>>(W2, W2h, DIM * DIM);

  // ======== Layer 1: GAT(heads=2, ch=64, concat), relu ========
  gemm128_wmma<<<NNODES / 16, 256, 0, stream>>>(Xh, W1h, A);
  alpha_kernel<<<cdiv((long long)NNODES * 2, T), T, 0, stream>>>(A, as1, ad1, asrc, adst, NNODES, 2, 64);
  fill_kernel<<<cdiv((long long)NNODES * 2, T), T, 0, stream>>>(emax, NEG_INF, NNODES * 2);
  fill_kernel<<<cdiv((long long)NNODES * 2, T), T, 0, stream>>>(denom, 0.f, NNODES * 2);
  fill_kernel<<<cdiv((long long)NNODES * DIM, T), T, 0, stream>>>(B, 0.f, NNODES * DIM);
  edge_max_kernel<<<cdiv((long long)EL * 2, T), T, 0, stream>>>(ei, asrc, adst, emax, NEDGES, NNODES, 2);
  edge_expsum_kernel<<<cdiv((long long)EL * 2, T), T, 0, stream>>>(ei, asrc, adst, emax, ee, denom, NEDGES, NNODES, 2);
  edge_aggregate_kernel<<<cdiv((long long)EL * 32, T), T, 0, stream>>>(ei, A, ee, denom, B, NEDGES, NNODES, 64, 2);
  bias_relu_f16_kernel<<<cdiv((long long)NNODES * DIM, T), T, 0, stream>>>(B, b1, Xh, NNODES * DIM);

  // ======== Layer 2: GAT(heads=1, ch=128), no relu ========
  gemm128_wmma<<<NNODES / 16, 256, 0, stream>>>(Xh, W2h, A);
  alpha_kernel<<<cdiv((long long)NNODES, T), T, 0, stream>>>(A, as2, ad2, asrc, adst, NNODES, 1, 128);
  fill_kernel<<<cdiv((long long)NNODES, T), T, 0, stream>>>(emax, NEG_INF, NNODES);
  fill_kernel<<<cdiv((long long)NNODES, T), T, 0, stream>>>(denom, 0.f, NNODES);
  fill_kernel<<<cdiv((long long)NNODES * DIM, T), T, 0, stream>>>(B, 0.f, NNODES * DIM);
  edge_max_kernel<<<cdiv((long long)EL, T), T, 0, stream>>>(ei, asrc, adst, emax, NEDGES, NNODES, 1);
  edge_expsum_kernel<<<cdiv((long long)EL, T), T, 0, stream>>>(ei, asrc, adst, emax, ee, denom, NEDGES, NNODES, 1);
  edge_aggregate_kernel<<<cdiv((long long)EL * 32, T), T, 0, stream>>>(ei, A, ee, denom, B, NEDGES, NNODES, 128, 1);
  bias_kernel<<<cdiv((long long)NNODES * DIM, T), T, 0, stream>>>(B, b2, NNODES * DIM);

  // ======== SimpleConv(mean) over raw edges + residual + relu ========
  fill_kernel<<<cdiv((long long)NNODES * DIM, T), T, 0, stream>>>(A, 0.f, NNODES * DIM);
  fill_kernel<<<cdiv((long long)NNODES, T), T, 0, stream>>>(cnt, 0.f, NNODES);
  edge_cnt_kernel<<<cdiv((long long)NEDGES, T), T, 0, stream>>>(ei, cnt, NEDGES);
  edge_sum_kernel<<<cdiv((long long)NEDGES * 32, T), T, 0, stream>>>(ei, B, A, NEDGES);
  final_kernel<<<cdiv((long long)NNODES * DIM, T), T, 0, stream>>>(A, cnt, x, out, NNODES);
}
